// PGCN_MIL_69097433858262
// MI455X (gfx1250) — matrix operand order repaired and checked by reading, not compiled, not run on hardware
//
#include <hip/hip_runtime.h>
#include <hip/hip_bf16.h>
#include <stdint.h>

// ---------------------------------------------------------------------------
// Problem constants (match reference)
// ---------------------------------------------------------------------------
#define NN_      16384
#define IN_DIM_  1024
#define HID_     256
#define NCLS_    2
#define KNN_     5

// ---------------------------------------------------------------------------
// CDNA5 WMMA types
// ---------------------------------------------------------------------------
typedef __attribute__((ext_vector_type(16))) __bf16        v16bf;
typedef __attribute__((ext_vector_type(8)))  float         v8f;
typedef __attribute__((ext_vector_type(8)))  unsigned int  v8u;

__device__ __forceinline__ unsigned short f32_to_bf16u(float f) {
    unsigned int u = __builtin_bit_cast(unsigned int, f);
    u += 0x7FFFu + ((u >> 16) & 1u);           // round-to-nearest-even
    return (unsigned short)(u >> 16);
}

// A-fragment (16x32 bf16, ISA 7.12.2): lane = {half = lane>>4, r = lane&15}
// row m = tileM + r; VGPR v<4 : K = kBase + half*8 + 2v   (+0,+1)
//                    VGPR v>=4: K = kBase + 16 + half*8 + 2(v-4)
__device__ __forceinline__ v16bf load_frag_a(const unsigned short* A, int lda,
                                             int m, int kBase, int half) {
    const unsigned short* p = A + (size_t)m * lda + kBase;
    v8u w;
    const int h8 = half * 8;
#pragma unroll
    for (int v = 0; v < 4; ++v)
        w[v] = *(const unsigned int*)(p + h8 + 2 * v);
#pragma unroll
    for (int v = 4; v < 8; ++v)
        w[v] = *(const unsigned int*)(p + 16 + h8 + 2 * (v - 4));
    return __builtin_bit_cast(v16bf, w);
}

// B-fragment (32x16 bf16) where B is stored transposed as rows of K
// (Bt[n*ldb + k] == B[k,n]): VGPR v<4: K = half*8 + 2v; v>=4: K = 16 + half*8 + 2(v-4).
__device__ __forceinline__ v16bf load_frag_b_rowK(const unsigned short* Bt, int ldb,
                                                  int n, int kBase, int half) {
    const unsigned short* p = Bt + (size_t)n * ldb + kBase + half * 8;
    v8u w;
#pragma unroll
    for (int v = 0; v < 8; ++v)
        w[v] = *(const unsigned int*)(p + (v >> 2) * 16 + 2 * (v & 3));
    return __builtin_bit_cast(v16bf, w);
}

__device__ __forceinline__ v8f wmma_bf16(v16bf a, v16bf b, v8f c) {
    return __builtin_amdgcn_wmma_f32_16x16x32_bf16(
        /*neg_a=*/false, a, /*neg_b=*/false, b,
        /*c_mod=*/(short)0, c, /*reuse_a=*/false, /*reuse_b=*/false);
}

// ---------------------------------------------------------------------------
// Conversion kernels
// ---------------------------------------------------------------------------
__global__ void cvt_bf16_k(const float* __restrict__ src,
                           unsigned short* __restrict__ dst, size_t n) {
    size_t i = (size_t)blockIdx.x * blockDim.x + threadIdx.x;
    size_t stride = (size_t)gridDim.x * blockDim.x;
    for (; i < n; i += stride) dst[i] = f32_to_bf16u(src[i]);
}

// dst[c*R + r] = bf16(src[r*C + c])  (weights -> [N,K] so B frags are contiguous)
__global__ void cvt_bf16_T_k(const float* __restrict__ src,
                             unsigned short* __restrict__ dst, int R, int C) {
    int i = blockIdx.x * blockDim.x + threadIdx.x;
    int total = R * C;
    if (i >= total) return;
    int r = i / C, c = i % C;
    dst[(size_t)c * R + r] = f32_to_bf16u(src[i]);
}

__global__ void zero_f32_k(float* __restrict__ p, size_t n) {
    size_t i = (size_t)blockIdx.x * blockDim.x + threadIdx.x;
    size_t stride = (size_t)gridDim.x * blockDim.x;
    for (; i < n; i += stride) p[i] = 0.0f;
}

// ---------------------------------------------------------------------------
// Tiled bf16 WMMA GEMM: C[M,Nc] = act(A[M,K] @ B[K,Nc] + bias)
// A row-major bf16, B supplied transposed [Nc,K] bf16. One wave per 16x16 tile.
// Optionally also emits bf16 copy of C.
// ---------------------------------------------------------------------------
__global__ __launch_bounds__(256)
void gemm_bf16_k(const unsigned short* __restrict__ A,
                 const unsigned short* __restrict__ Bt,
                 const float* __restrict__ bias,
                 float* __restrict__ C, unsigned short* __restrict__ Cb,
                 int M, int Nc, int Kk, int relu) {
    const int wave  = threadIdx.x >> 5;
    const int lane  = threadIdx.x & 31;
    const int tile  = blockIdx.x * (blockDim.x >> 5) + wave;
    const int tilesN = Nc >> 4;
    const int tm = (tile / tilesN) << 4;
    const int tn = (tile % tilesN) << 4;
    if (tm >= M) return;                    // wave-uniform exit (EXEC stays all-1)
    const int half = lane >> 4, r = lane & 15;

    v8f c = {};
    for (int kb = 0; kb < Kk; kb += 32) {
        v16bf a = load_frag_a(A, Kk, tm + r, kb, half);
        v16bf b = load_frag_b_rowK(Bt, Kk, tn + r, kb, half);
        c = wmma_bf16(a, b, c);
    }
#pragma unroll
    for (int v = 0; v < 8; ++v) {
        const int row = tm + v + half * 8;
        const int col = tn + r;
        float val = c[v] + (bias ? bias[col] : 0.0f);
        if (relu) val = fmaxf(val, 0.0f);
        C[(size_t)row * Nc + col] = val;
        if (Cb) Cb[(size_t)row * Nc + col] = f32_to_bf16u(val);
    }
}

// ---------------------------------------------------------------------------
// Row squared norms of h (f32), one wave per row (HID=256 -> 8 elems/lane)
// ---------------------------------------------------------------------------
__global__ __launch_bounds__(256)
void sqnorm_k(const float* __restrict__ h, float* __restrict__ sq, int N) {
    const int wave = threadIdx.x >> 5, lane = threadIdx.x & 31;
    const int row  = blockIdx.x * 8 + wave;
    if (row >= N) return;
    float acc = 0.0f;
#pragma unroll
    for (int t = lane; t < HID_; t += 32) {
        float v = h[(size_t)row * HID_ + t];
        acc += v * v;
    }
#pragma unroll
    for (int o = 16; o > 0; o >>= 1) acc += __shfl_down(acc, o, 32);
    if (lane == 0) sq[row] = acc;
}

// ---------------------------------------------------------------------------
// Fused kNN: one block per 32-row tile (two A-fragment sets per wave, each
// streamed B-fragment feeds TWO wmma -> halves L2 B-traffic vs 16-row tiles).
// 8 waves stream the 1024 column tiles round-robin. d2 = sqi + sqj - 2*dot,
// diagonal forced to -1 (self). Per-lane running top-6 for row = lane.
// ---------------------------------------------------------------------------
#define KNN_WAVES 8
#define KNN_ROWS  32
__global__ __launch_bounds__(256)
void knn_topk_k(const unsigned short* __restrict__ Hb,
                const float* __restrict__ sq,
                int* __restrict__ nbr, int N) {
    __shared__ float tileLds[KNN_WAVES][KNN_ROWS][17];
    __shared__ float candD[KNN_WAVES][KNN_ROWS][6];
    __shared__ int   candI[KNN_WAVES][KNN_ROWS][6];

    const int wave = threadIdx.x >> 5, lane = threadIdx.x & 31;
    const int half = lane >> 4, r = lane & 15;
    const int tm = blockIdx.x * KNN_ROWS;

    // Preload both 16-row A tiles (K=256 -> 2 x 8 fragments) into VGPRs.
    v16bf afr0[8], afr1[8];
#pragma unroll
    for (int s = 0; s < 8; ++s) {
        afr0[s] = load_frag_a(Hb, HID_, tm + r,      s * 32, half);
        afr1[s] = load_frag_a(Hb, HID_, tm + 16 + r, s * 32, half);
    }
    float srow0[8], srow1[8];
#pragma unroll
    for (int v = 0; v < 8; ++v) {
        srow0[v] = sq[tm + v + half * 8];
        srow1[v] = sq[tm + 16 + v + half * 8];
    }

    float bestD[6]; int bestI[6];
#pragma unroll
    for (int q = 0; q < 6; ++q) { bestD[q] = 3.4e38f; bestI[q] = -1; }

    for (int tj = wave; tj < (N >> 4); tj += KNN_WAVES) {
        const int tn = tj * 16;
        __builtin_prefetch(Hb + (size_t)(tn + 16 * KNN_WAVES + r) * HID_, 0, 1);
        v8f c0 = {}, c1 = {};
#pragma unroll
        for (int s = 0; s < 8; ++s) {
            v16bf b = load_frag_b_rowK(Hb, HID_, tn + r, s * 32, half);
            c0 = wmma_bf16(afr0[s], b, c0);     // B fragment reused twice
            c1 = wmma_bf16(afr1[s], b, c1);
        }
        const float sqn = sq[tn + r];
#pragma unroll
        for (int v = 0; v < 8; ++v) {
            const int row0 = v + half * 8;
            const int row1 = 16 + v + half * 8;
            float d0 = srow0[v] + sqn - 2.0f * c0[v];
            float d1 = srow1[v] + sqn - 2.0f * c1[v];
            if (tm + row0 == tn + r) d0 = -1.0f;  // self
            if (tm + row1 == tn + r) d1 = -1.0f;
            tileLds[wave][row0][r] = d0;          // intra-wave LDS bounce
            tileLds[wave][row1][r] = d1;
        }
        // all 32 lanes: lane owns row `lane` of the 32-row tile
#pragma unroll
        for (int n = 0; n < 16; ++n) {
            float dd = tileLds[wave][lane][n];
            int   ii = tn + n;
            if (dd < bestD[5]) {
#pragma unroll
                for (int q = 0; q < 6; ++q) {
                    if (dd < bestD[q]) {
                        float td = bestD[q]; int ti = bestI[q];
                        bestD[q] = dd; bestI[q] = ii; dd = td; ii = ti;
                    }
                }
            }
        }
    }
#pragma unroll
    for (int q = 0; q < 6; ++q) {
        candD[wave][lane][q] = bestD[q];
        candI[wave][lane][q] = bestI[q];
    }
    __syncthreads();
    if (threadIdx.x < KNN_ROWS) {
        const int row = threadIdx.x;
        float cd[KNN_WAVES * 6]; int ci[KNN_WAVES * 6];
#pragma unroll
        for (int w = 0; w < KNN_WAVES; ++w)
#pragma unroll
            for (int q = 0; q < 6; ++q) {
                cd[w * 6 + q] = candD[w][row][q];
                ci[w * 6 + q] = candI[w][row][q];
            }
        for (int rank = 0; rank < 6; ++rank) {
            int best = 0; float bd = cd[0];
            for (int e = 1; e < KNN_WAVES * 6; ++e)
                if (cd[e] < bd) { bd = cd[e]; best = e; }
            if (rank > 0) nbr[(size_t)(tm + row) * KNN_ + (rank - 1)] = ci[best];
            cd[best] = 3.4e38f;
        }
    }
}

// ---------------------------------------------------------------------------
// Graph normalization: deg[i] = K+1 + (#times i appears in nbr); dinv = deg^-1/2
// ---------------------------------------------------------------------------
__global__ void deg_init_k(float* __restrict__ deg, int N) {
    int i = blockIdx.x * blockDim.x + threadIdx.x;
    if (i < N) deg[i] = (float)(KNN_ + 1);
}
__global__ void deg_scatter_k(const int* __restrict__ nbr, float* __restrict__ deg,
                              int total) {
    int e = blockIdx.x * blockDim.x + threadIdx.x;
    if (e < total) atomicAdd(&deg[nbr[e]], 1.0f);
}
__global__ void dinv_k(float* __restrict__ deg, int N) {
    int i = blockIdx.x * blockDim.x + threadIdx.x;
    if (i < N) { float d = deg[i]; deg[i] = (d > 0.0f) ? rsqrtf(d) : 0.0f; }
}

// ---------------------------------------------------------------------------
// Symmetric-normalized GCN aggregation (self loop + both edge directions).
// One block per node, thread = feature dim. agg must be pre-zeroed.
// ---------------------------------------------------------------------------
__global__ __launch_bounds__(HID_)
void gcn_agg_scatter_k(const float* __restrict__ hW, const int* __restrict__ nbr,
                       const float* __restrict__ dinv, float* __restrict__ agg) {
    const int i = blockIdx.x;
    const int t = threadIdx.x;
    const float di = dinv[i];
    const float hs = hW[(size_t)i * HID_ + t] * di;   // hW[i]*dinv[i]
    int nb[KNN_]; float dj[KNN_];
#pragma unroll
    for (int k = 0; k < KNN_; ++k) { nb[k] = nbr[(size_t)i * KNN_ + k]; dj[k] = dinv[nb[k]]; }
    float acc = hs * di;                              // self loop
#pragma unroll
    for (int k = 0; k < KNN_; ++k)                    // incoming j -> i
        acc += hW[(size_t)nb[k] * HID_ + t] * dj[k] * di;
    atomicAdd(&agg[(size_t)i * HID_ + t], acc);
#pragma unroll
    for (int k = 0; k < KNN_; ++k)                    // outgoing i -> j
        atomicAdd(&agg[(size_t)nb[k] * HID_ + t], hs * dj[k]);
}

__global__ void gcn_finalize_k(const float* __restrict__ agg,
                               const float* __restrict__ bias,
                               float* __restrict__ Hc, unsigned short* __restrict__ Hcb,
                               int relu, size_t total) {
    size_t i = (size_t)blockIdx.x * blockDim.x + threadIdx.x;
    size_t stride = (size_t)gridDim.x * blockDim.x;
    for (; i < total; i += stride) {
        float v = agg[i] + bias[i % HID_];
        if (relu) v = fmaxf(v, 0.0f);
        Hc[i] = v;
        if (Hcb) Hcb[i] = f32_to_bf16u(v);
    }
}

// ---------------------------------------------------------------------------
// Mean pool + classifier
// ---------------------------------------------------------------------------
__global__ __launch_bounds__(HID_)
void colsum_k(const float* __restrict__ h2, float* __restrict__ bag, int N) {
    const int t = threadIdx.x;
    float acc = 0.0f;
    for (int row = blockIdx.x; row < N; row += gridDim.x)
        acc += h2[(size_t)row * HID_ + t];
    atomicAdd(&bag[t], acc);
}

__global__ __launch_bounds__(32)
void cls_k(const float* __restrict__ bag, const float* __restrict__ Wcls,
           const float* __restrict__ bcls, float* __restrict__ out, int N) {
    const int c = blockIdx.x;           // 2 classes
    const int lane = threadIdx.x;
    const float invN = 1.0f / (float)N;
    float acc = 0.0f;
    for (int t = lane; t < HID_; t += 32)
        acc += (bag[t] * invN) * Wcls[(size_t)t * NCLS_ + c];
#pragma unroll
    for (int o = 16; o > 0; o >>= 1) acc += __shfl_down(acc, o, 32);
    if (lane == 0) out[c] = acc + bcls[c];
}

// ---------------------------------------------------------------------------
// Host launcher
// ---------------------------------------------------------------------------
extern "C" void kernel_launch(void* const* d_in, const int* in_sizes, int n_in,
                              void* d_out, int out_size, void* d_ws, size_t ws_size,
                              hipStream_t stream) {
    const float* x      = (const float*)d_in[0];
    const float* Wproj  = (const float*)d_in[1];
    const float* bproj  = (const float*)d_in[2];
    const float* Wg1    = (const float*)d_in[3];
    const float* bg1    = (const float*)d_in[4];
    const float* Wg2    = (const float*)d_in[5];
    const float* bg2    = (const float*)d_in[6];
    const float* Wcls   = (const float*)d_in[7];
    const float* bcls   = (const float*)d_in[8];
    float* out = (float*)d_out;

    char* w = (char*)d_ws;
    size_t off = 0;
    auto alloc = [&](size_t bytes) { size_t o = off; off += (bytes + 255) & ~(size_t)255; return o; };

    // Big region: xb (32 MB) reused later as hW (16 MB) + agg (16 MB)
    const size_t o_big  = alloc((size_t)NN_ * IN_DIM_ * 2);
    const size_t o_h    = alloc((size_t)NN_ * HID_ * 4);
    const size_t o_hb   = alloc((size_t)NN_ * HID_ * 2);
    const size_t o_WpT  = alloc((size_t)IN_DIM_ * HID_ * 2);
    const size_t o_W1T  = alloc((size_t)HID_ * HID_ * 2);
    const size_t o_W2T  = alloc((size_t)HID_ * HID_ * 2);
    const size_t o_sq   = alloc((size_t)NN_ * 4);
    const size_t o_dinv = alloc((size_t)NN_ * 4);
    const size_t o_nbr  = alloc((size_t)NN_ * KNN_ * 4);
    const size_t o_bag  = alloc((size_t)HID_ * 4);
    (void)ws_size; (void)n_in; (void)in_sizes; (void)out_size;

    unsigned short* xb   = (unsigned short*)(w + o_big);
    float*          hW   = (float*)(w + o_big);                          // aliases xb
    float*          agg  = (float*)(w + o_big + (size_t)NN_ * HID_ * 4); // aliases xb
    float*          h    = (float*)(w + o_h);
    unsigned short* hb   = (unsigned short*)(w + o_hb);
    unsigned short* WpT  = (unsigned short*)(w + o_WpT);
    unsigned short* W1T  = (unsigned short*)(w + o_W1T);
    unsigned short* W2T  = (unsigned short*)(w + o_W2T);
    float*          sq   = (float*)(w + o_sq);
    float*          dinv = (float*)(w + o_dinv);
    int*            nbr  = (int*)(w + o_nbr);
    float*          bag  = (float*)(w + o_bag);

    const size_t nAgg = (size_t)NN_ * HID_;

    // 1) bf16 conversions (weights transposed to [N,K])
    cvt_bf16_k<<<2048, 256, 0, stream>>>(x, xb, (size_t)NN_ * IN_DIM_);
    cvt_bf16_T_k<<<(IN_DIM_ * HID_ + 255) / 256, 256, 0, stream>>>(Wproj, WpT, IN_DIM_, HID_);
    cvt_bf16_T_k<<<(HID_ * HID_ + 255) / 256, 256, 0, stream>>>(Wg1, W1T, HID_, HID_);
    cvt_bf16_T_k<<<(HID_ * HID_ + 255) / 256, 256, 0, stream>>>(Wg2, W2T, HID_, HID_);

    // 2) h = relu(x @ Wproj + bproj)  [WMMA], also bf16 copy
    {
        int tiles = (NN_ / 16) * (HID_ / 16);
        gemm_bf16_k<<<tiles / 8, 256, 0, stream>>>(xb, WpT, bproj, h, hb,
                                                   NN_, HID_, IN_DIM_, 1);
    }

    // 3) row norms + fused kNN top-k [WMMA]
    sqnorm_k<<<NN_ / 8, 256, 0, stream>>>(h, sq, NN_);
    knn_topk_k<<<NN_ / KNN_ROWS, 256, 0, stream>>>(hb, sq, nbr, NN_);

    // 4) degrees -> dinv
    deg_init_k<<<(NN_ + 255) / 256, 256, 0, stream>>>(dinv, NN_);
    deg_scatter_k<<<(NN_ * KNN_ + 255) / 256, 256, 0, stream>>>(nbr, dinv, NN_ * KNN_);
    dinv_k<<<(NN_ + 255) / 256, 256, 0, stream>>>(dinv, NN_);

    // 5) GCN layer 1: hW = h @ Wg1 [WMMA]; aggregate; relu(+b)
    {
        int tiles = (NN_ / 16) * (HID_ / 16);
        gemm_bf16_k<<<tiles / 8, 256, 0, stream>>>(hb, W1T, nullptr, hW, nullptr,
                                                   NN_, HID_, HID_, 0);
    }
    zero_f32_k<<<2048, 256, 0, stream>>>(agg, nAgg);
    gcn_agg_scatter_k<<<NN_, HID_, 0, stream>>>(hW, nbr, dinv, agg);
    gcn_finalize_k<<<2048, 256, 0, stream>>>(agg, bg1, h, hb, 1, nAgg);  // h<-h1, hb<-h1b

    // 6) GCN layer 2
    {
        int tiles = (NN_ / 16) * (HID_ / 16);
        gemm_bf16_k<<<tiles / 8, 256, 0, stream>>>(hb, W2T, nullptr, hW, nullptr,
                                                   NN_, HID_, HID_, 0);
    }
    zero_f32_k<<<2048, 256, 0, stream>>>(agg, nAgg);
    gcn_agg_scatter_k<<<NN_, HID_, 0, stream>>>(hW, nbr, dinv, agg);
    gcn_finalize_k<<<2048, 256, 0, stream>>>(agg, bg2, h, nullptr, 0, nAgg); // h<-h2

    // 7) mean pool + classifier
    zero_f32_k<<<1, 256, 0, stream>>>(bag, (size_t)HID_);
    colsum_k<<<128, HID_, 0, stream>>>(h, bag, NN_);
    cls_k<<<NCLS_, 32, 0, stream>>>(bag, Wcls, bcls, out, NN_);
}
